// LstmBlock_32117765439866
// MI455X (gfx1250) — matrix operand order repaired and checked by reading, not compiled, not run on hardware
//
#include <hip/hip_runtime.h>
#include <hip/hip_bf16.h>
#include <cstdint>
#include <cstddef>

// LSTM block for MI455X (gfx1250, wave32, WMMA).
//   B=64, T=256, D=512, H=1024
// Plan:
//   1) prep: x -> bf16, W,R -> transposed bf16 ([N][K] so B-fragments are b128 loads)
//   2) Z = x@W + bias  (fp32, parallel WMMA GEMM, off the critical path)
//   3) 256 sequential step kernels: z_t + h@R via v_wmma_f32_16x16x32_bf16,
//      K split over 4 waves + LDS reduce, fused gate math, h double-buffered bf16
//   4) BN + LeakyReLU in place on d_out

#define BB 64
#define TT 256
#define DD 512
#define HH 1024
#define G4 (4 * HH)

typedef __attribute__((ext_vector_type(16))) __bf16 v16bf;
typedef __attribute__((ext_vector_type(8)))  float  v8f;

union FragAB {           // one WMMA A or B operand: 16 bf16 per lane = 2 x b128
    v16bf v;
    uint4 q[2];
};

__device__ __forceinline__ float sigf(float x) { return 1.0f / (1.0f + __expf(-x)); }

// ---------------------------------------------------------------- prep kernels
__global__ void k_cvt_bf16(const float* __restrict__ src, __bf16* __restrict__ dst, int n) {
    int i = blockIdx.x * blockDim.x + threadIdx.x;
    if (i < n) dst[i] = (__bf16)src[i];
}

// src: [K][N] fp32 row-major  ->  dst: [N][K] bf16 row-major (LDS-tiled transpose)
__global__ void k_transpose_to_bf16(const float* __restrict__ src, __bf16* __restrict__ dst,
                                    int K, int N) {
    __shared__ float tile[32][33];
    const int kb = blockIdx.y * 32, nb = blockIdx.x * 32;
    const int tx = threadIdx.x, ty = threadIdx.y;   // block (32, 8)
#pragma unroll
    for (int i = 0; i < 32; i += 8)
        tile[ty + i][tx] = src[(size_t)(kb + ty + i) * N + nb + tx];
    __syncthreads();
#pragma unroll
    for (int i = 0; i < 32; i += 8)
        dst[(size_t)(nb + ty + i) * K + kb + tx] = (__bf16)tile[tx][ty + i];
}

__global__ void k_init_state(float* __restrict__ c, __bf16* __restrict__ h0,
                             __bf16* __restrict__ h1) {
    int i = blockIdx.x * blockDim.x + threadIdx.x;  // BB*HH threads
    c[i] = 0.0f;
    h0[i] = (__bf16)0.0f;
    h1[i] = (__bf16)0.0f;
}

// ------------------------------------------------- phase 1: Z = x@W + bias (fp32)
// One wave per block computes a 64(M) x 64(N) tile, K = DD = 512.
__global__ void __launch_bounds__(32)
k_gemm_z(const __bf16* __restrict__ Xb,   // [B*T][D] bf16
         const __bf16* __restrict__ Wt,   // [4H][D] bf16 (W transposed)
         const float*  __restrict__ bias, // [4H]
         float* __restrict__ Z)           // [B*T][4H] fp32
{
    const int lane  = threadIdx.x & 31;
    const int ln15  = lane & 15;
    const int hi    = lane >> 4;
    const int kA    = hi * 8;    // A: lane-half picks K {0..7,16..23} vs {8..15,24..31}
    const int kB    = hi * 8;    // B: lane-half picks K {0..7,16..23} vs {8..15,24..31}
    const int m_base = blockIdx.y * 64;
    const int n_base = blockIdx.x * 64;

    const v8f vzero = {0.f, 0.f, 0.f, 0.f, 0.f, 0.f, 0.f, 0.f};
    v8f acc[4][4];
#pragma unroll
    for (int mi = 0; mi < 4; ++mi)
#pragma unroll
        for (int ni = 0; ni < 4; ++ni) acc[mi][ni] = vzero;

    for (int k0 = 0; k0 < DD; k0 += 32) {
        FragAB A[4], Bf[4];
#pragma unroll
        for (int mi = 0; mi < 4; ++mi) {
            const __bf16* p = Xb + (size_t)(m_base + mi * 16 + ln15) * DD + k0 + kA;
            A[mi].q[0] = *(const uint4*)p;
            A[mi].q[1] = *(const uint4*)(p + 16);
        }
#pragma unroll
        for (int ni = 0; ni < 4; ++ni) {
            const __bf16* p = Wt + (size_t)(n_base + ni * 16 + ln15) * DD + k0 + kB;
            Bf[ni].q[0] = *(const uint4*)p;
            Bf[ni].q[1] = *(const uint4*)(p + 16);
        }
#pragma unroll
        for (int mi = 0; mi < 4; ++mi)
#pragma unroll
            for (int ni = 0; ni < 4; ++ni)
                acc[mi][ni] = __builtin_amdgcn_wmma_f32_16x16x32_bf16(
                    false, A[mi].v, false, Bf[ni].v, (short)0, acc[mi][ni], false, false);
    }

#pragma unroll
    for (int ni = 0; ni < 4; ++ni) {
        const int n = n_base + ni * 16 + ln15;
        const float bi = bias[n];
#pragma unroll
        for (int mi = 0; mi < 4; ++mi) {
            const int m = m_base + mi * 16 + hi * 8;
#pragma unroll
            for (int r = 0; r < 8; ++r)
                Z[(size_t)(m + r) * G4 + n] = acc[mi][ni][r] + bi;
        }
    }
}

// ------------------------------------------- phase 2: one LSTM timestep (x256)
// Block = 4 waves, handles all M=64 rows x one 16-wide h-column tile x 4 gates.
// Each wave accumulates a K-chunk of 256; LDS reduce; wave w applies gates for
// M-tile mi == w. h is double-buffered (read-all / written-by-owner).
__global__ void __launch_bounds__(128)
k_step(const __bf16* __restrict__ h_prev, // [B][H] bf16
       __bf16* __restrict__ h_next,       // [B][H] bf16
       float*  __restrict__ c,            // [B][H] fp32 (in/out)
       const float* __restrict__ Z,       // [B*T][4H] fp32 (bias included)
       const __bf16* __restrict__ Rt,     // [4H][H] bf16 (rec_kernel transposed)
       float* __restrict__ out,           // [B][T][H] fp32 (raw h, pre-BN)
       int t)
{
    __shared__ float red[4][4][4][8][32]; // [wave][mi][gate][row][lane] = 64 KB
    const int lane = threadIdx.x & 31;
    const int wv   = threadIdx.x >> 5;
    const int ln15 = lane & 15;
    const int hi   = lane >> 4;
    const int n0   = blockIdx.x * 16;

    const v8f vzero = {0.f, 0.f, 0.f, 0.f, 0.f, 0.f, 0.f, 0.f};
    v8f acc[4][4];                        // [mi][gate]
#pragma unroll
    for (int mi = 0; mi < 4; ++mi)
#pragma unroll
        for (int g = 0; g < 4; ++g) acc[mi][g] = vzero;

    const int kbeg = wv * 256, kend = kbeg + 256;
    for (int k0 = kbeg; k0 < kend; k0 += 32) {
        FragAB A[4], Bf[4];
#pragma unroll
        for (int mi = 0; mi < 4; ++mi) {
            const __bf16* p = h_prev + (size_t)(mi * 16 + ln15) * HH + k0 + hi * 8;
            A[mi].q[0] = *(const uint4*)p;
            A[mi].q[1] = *(const uint4*)(p + 16);
        }
#pragma unroll
        for (int g = 0; g < 4; ++g) {
            const __bf16* p = Rt + (size_t)(g * HH + n0 + ln15) * HH + k0 + hi * 8;
            Bf[g].q[0] = *(const uint4*)p;
            Bf[g].q[1] = *(const uint4*)(p + 16);
            if (k0 + 32 < kend)           // hint next K-slab of weights toward L0/L2
                __builtin_prefetch((const void*)(p + 32), 0, 1);
        }
#pragma unroll
        for (int mi = 0; mi < 4; ++mi)
#pragma unroll
            for (int g = 0; g < 4; ++g)
                acc[mi][g] = __builtin_amdgcn_wmma_f32_16x16x32_bf16(
                    false, A[mi].v, false, Bf[g].v, (short)0, acc[mi][g], false, false);
    }

#pragma unroll
    for (int mi = 0; mi < 4; ++mi)
#pragma unroll
        for (int g = 0; g < 4; ++g)
#pragma unroll
            for (int r = 0; r < 8; ++r)
                red[wv][mi][g][r][lane] = acc[mi][g][r];
    __syncthreads();

    const int mi = wv;                    // this wave finalizes M-tile 'wv'
#pragma unroll
    for (int r = 0; r < 8; ++r) {
        const int b = mi * 16 + hi * 8 + r;
        const int n = n0 + ln15;
        const size_t zrow = ((size_t)b * TT + t) * G4 + n;
        const float vi = red[0][mi][0][r][lane] + red[1][mi][0][r][lane] +
                         red[2][mi][0][r][lane] + red[3][mi][0][r][lane] + Z[zrow];
        const float vf = red[0][mi][1][r][lane] + red[1][mi][1][r][lane] +
                         red[2][mi][1][r][lane] + red[3][mi][1][r][lane] + Z[zrow + HH];
        const float vg = red[0][mi][2][r][lane] + red[1][mi][2][r][lane] +
                         red[2][mi][2][r][lane] + red[3][mi][2][r][lane] + Z[zrow + 2 * HH];
        const float vo = red[0][mi][3][r][lane] + red[1][mi][3][r][lane] +
                         red[2][mi][3][r][lane] + red[3][mi][3][r][lane] + Z[zrow + 3 * HH];
        const float ig = sigf(vi);
        const float fg = sigf(vf);
        const float gg = tanhf(vg);
        const float og = sigf(vo);
        const int ci = b * HH + n;
        const float cn = fg * c[ci] + ig * gg;
        c[ci] = cn;
        const float hv = og * tanhf(cn);
        out[((size_t)b * TT + t) * HH + n] = hv;
        h_next[ci] = (__bf16)hv;
    }
}

// ------------------------------------------ phase 3: BatchNorm + LeakyReLU(0.3)
__global__ void k_bn_lrelu(float* __restrict__ y, const float* __restrict__ gamma,
                           const float* __restrict__ beta, const float* __restrict__ mean,
                           const float* __restrict__ var) {
    const size_t i = (size_t)blockIdx.x * blockDim.x + threadIdx.x;
    const int n = (int)(i & (HH - 1));
    const float inv = gamma[n] * rsqrtf(var[n] + 1e-3f);
    float v = (y[i] - mean[n]) * inv + beta[n];
    y[i] = (v >= 0.0f) ? v : 0.3f * v;
}

// ------------------------------------------------------------------ launcher
extern "C" void kernel_launch(void* const* d_in, const int* in_sizes, int n_in,
                              void* d_out, int out_size, void* d_ws, size_t ws_size,
                              hipStream_t stream) {
    (void)in_sizes; (void)n_in; (void)out_size; (void)ws_size;

    const float* x     = (const float*)d_in[0];  // [B,T,D]
    const float* W     = (const float*)d_in[1];  // [D,4H]
    const float* R     = (const float*)d_in[2];  // [H,4H]
    const float* bias  = (const float*)d_in[3];  // [4H]
    const float* gamma = (const float*)d_in[4];  // [H]
    const float* beta  = (const float*)d_in[5];  // [H]
    const float* mmean = (const float*)d_in[6];  // [H]
    const float* mvar  = (const float*)d_in[7];  // [H]
    float* out = (float*)d_out;                  // [B,T,H]

    char* ws = (char*)d_ws;
    size_t off = 0;
    float*  Z  = (float*)(ws + off);  off += (size_t)BB * TT * G4 * sizeof(float);  // 256 MiB
    __bf16* Rt = (__bf16*)(ws + off); off += (size_t)G4 * HH * sizeof(__bf16);      //   8 MiB
    __bf16* Wt = (__bf16*)(ws + off); off += (size_t)G4 * DD * sizeof(__bf16);      //   4 MiB
    __bf16* Xb = (__bf16*)(ws + off); off += (size_t)BB * TT * DD * sizeof(__bf16); //  16 MiB
    float*  cS = (float*)(ws + off);  off += (size_t)BB * HH * sizeof(float);
    __bf16* h0 = (__bf16*)(ws + off); off += (size_t)BB * HH * sizeof(__bf16);
    __bf16* h1 = (__bf16*)(ws + off); off += (size_t)BB * HH * sizeof(__bf16);

    // prep: conversions, weight transposes, zero state
    {
        const int n = BB * TT * DD;
        k_cvt_bf16<<<(n + 255) / 256, 256, 0, stream>>>(x, Xb, n);
    }
    k_transpose_to_bf16<<<dim3(G4 / 32, DD / 32), dim3(32, 8), 0, stream>>>(W, Wt, DD, G4);
    k_transpose_to_bf16<<<dim3(G4 / 32, HH / 32), dim3(32, 8), 0, stream>>>(R, Rt, HH, G4);
    k_init_state<<<(BB * HH) / 256, 256, 0, stream>>>(cS, h0, h1);

    // phase 1: all-timestep input projection (parallel WMMA GEMM)
    k_gemm_z<<<dim3(G4 / 64, (BB * TT) / 64), 32, 0, stream>>>(Xb, Wt, bias, Z);

    // phase 2: sequential recurrence; stream ordering = inter-step sync
    for (int t = 0; t < TT; ++t) {
        const __bf16* hp = (t & 1) ? h1 : h0;
        __bf16*       hn = (t & 1) ? h0 : h1;
        k_step<<<HH / 16, 128, 0, stream>>>(hp, hn, cS, Z, Rt, out, t);
    }

    // phase 3: BN (inference stats) + LeakyReLU, in place on d_out
    k_bn_lrelu<<<(unsigned)((size_t)BB * TT * HH / 256), 256, 0, stream>>>(
        out, gamma, beta, mmean, mvar);
}
